// MergeDecoder_59433757442357
// MI455X (gfx1250) — matrix operand order repaired and checked by reading, not compiled, not run on hardware
//
#include <hip/hip_runtime.h>

typedef __attribute__((ext_vector_type(16))) _Float16 v16h;
typedef __attribute__((ext_vector_type(8)))  float    v8f;

#define D       256
#define TWO_D   512
#define TILE_M  32      // edges per wave (two 16-row WMMA tiles sharing B fragments)
#define KSTEPS  16      // 512 / 32
#define NTILES  16      // 256 / 16
#define NT_GROUP 4      // n-tiles staged in LDS at a time (64 KB)
#define WAVES   8
#define BLOCK   256

// frag_w1 layout: [nt][k][lane][h] f16 — B-fragment order for v_wmma_f32_16x16x32_f16.
// B 32x16 tile: lanes 0-15 = column N (hold K=0..7,16..23), lanes 16-31 = same column (K=8..15,24..31).
__global__ __launch_bounds__(256) void prep_w1(const float* __restrict__ w1,
                                               _Float16* __restrict__ frag) {
  int i = blockIdx.x * 256 + threadIdx.x;           // 16*16*32*16 = 131072 total
  int h    =  i        & 15;
  int lane = (i >> 4)  & 31;
  int k    = (i >> 9)  & 15;
  int nt   =  i >> 13;
  int n  = nt * 16 + (lane & 15);                   // output-feature column
  int kl = (h & 7) + ((h >> 3) << 4) + ((lane >> 4) << 3); // K within the 32-deep tile
  int kk = k * 32 + kl;                             // K within [0,512)
  frag[i] = (_Float16)w1[n * TWO_D + kk];           // B = w1^T  =>  B[kk][n] = w1[n][kk]
}

__device__ __forceinline__ void load_a_frags(const float* __restrict__ z,
                                             const int*   __restrict__ srcIdx,
                                             const int*   __restrict__ dstIdx,
                                             int e, int halfSel, v16h* a) {
  const long sOff = (long)srcIdx[e] * D;
  const long dOff = (long)dstIdx[e] * D;
  #pragma unroll
  for (int k = 0; k < KSTEPS; ++k) {
    const float* rp = (k < 8) ? (z + sOff) : (z + dOff);
    const int c0 = ((k & 7) << 5) + halfSel;
    const float4 f0 = *(const float4*)(rp + c0);
    const float4 f1 = *(const float4*)(rp + c0 + 4);
    const float4 f2 = *(const float4*)(rp + c0 + 16);
    const float4 f3 = *(const float4*)(rp + c0 + 20);
    v16h av;
    av[0] =(_Float16)f0.x; av[1] =(_Float16)f0.y; av[2] =(_Float16)f0.z; av[3] =(_Float16)f0.w;
    av[4] =(_Float16)f1.x; av[5] =(_Float16)f1.y; av[6] =(_Float16)f1.z; av[7] =(_Float16)f1.w;
    av[8] =(_Float16)f2.x; av[9] =(_Float16)f2.y; av[10]=(_Float16)f2.z; av[11]=(_Float16)f2.w;
    av[12]=(_Float16)f3.x; av[13]=(_Float16)f3.y; av[14]=(_Float16)f3.z; av[15]=(_Float16)f3.w;
    a[k] = av;
  }
}

__global__ __launch_bounds__(256) void edge_mlp(
    const float*    __restrict__ z,
    const int*      __restrict__ srcIdx,
    const int*      __restrict__ dstIdx,
    const _Float16* __restrict__ frag_w1,
    const float*    __restrict__ b1,
    const float*    __restrict__ w2,
    const float*    __restrict__ b2,
    float*          __restrict__ out)
{
  __shared__ __attribute__((aligned(32))) _Float16 lds_frag[NT_GROUP * KSTEPS * 32 * 16]; // 64 KB

  const int lane = threadIdx.x & 31;
  const int wave = threadIdx.x >> 5;
  const int tile = blockIdx.x * WAVES + wave;       // 32-edge macrotile; exact division

  // A-fragment gather: lane L and L+16 both serve edge-row L (different K halves).
  const int row     = lane & 15;
  const int halfSel = (lane >> 4) << 3;             // lanes>=16 take K+8 / K+24 sub-chunks
  const int e0      = tile * TILE_M + row;          // rows 0..15 of the macrotile
  const int e1      = e0 + 16;                      // rows 16..31

  v16h a0[KSTEPS], a1[KSTEPS];                      // 256 VGPRs of resident gathered A
  load_a_frags(z, srcIdx, dstIdx, e0, halfSel, a0);
  load_a_frags(z, srcIdx, dstIdx, e1, halfSel, a1);

  float p0[8], p1[8];
  #pragma unroll
  for (int r = 0; r < 8; ++r) { p0[r] = 0.f; p1[r] = 0.f; }

  const uint4* gw = (const uint4*)frag_w1;
  uint4*       lw = (uint4*)lds_frag;

  for (int g = 0; g < NTILES / NT_GROUP; ++g) {
    __syncthreads();                                // WAR: previous group fully consumed
    // stage 4 n-tiles of W1 fragments (64 KB) into LDS, fully coalesced b128s
    const uint4* gsrc = gw + g * (NT_GROUP * KSTEPS * 32 * 16 * 2 / 16);
    #pragma unroll
    for (int i = 0; i < 16; ++i)
      lw[threadIdx.x + i * BLOCK] = gsrc[threadIdx.x + i * BLOCK];
    __syncthreads();

    #pragma unroll
    for (int ntl = 0; ntl < NT_GROUP; ++ntl) {
      const int nt = g * NT_GROUP + ntl;
      const v16h* bf = (const v16h*)lds_frag + ntl * (KSTEPS * 32) + lane;
      v8f acc0 = {}, acc1 = {};
      #pragma unroll
      for (int k = 0; k < KSTEPS; ++k) {
        const v16h bv = bf[k * 32];                 // 1 KB LDS read feeds TWO wmmas
        acc0 = __builtin_amdgcn_wmma_f32_16x16x32_f16(false, a0[k], false, bv,
                                                      (short)0, acc0, false, false);
        acc1 = __builtin_amdgcn_wmma_f32_16x16x32_f16(false, a1[k], false, bv,
                                                      (short)0, acc1, false, false);
      }
      // fused layer 2: partial += relu(h + b1[n]) * w2[n]
      const int   n   = nt * 16 + row;              // N = lane%16 within this n-tile
      const float b1v = b1[n];
      const float w2v = w2[n];
      #pragma unroll
      for (int r = 0; r < 8; ++r) {
        float h0 = acc0[r] + b1v; h0 = h0 > 0.f ? h0 : 0.f; p0[r] += h0 * w2v;
        float h1 = acc1[r] + b1v; h1 = h1 > 0.f ? h1 : 0.f; p1[r] += h1 * w2v;
      }
    }
  }

  // reduce across the 16 lanes of each half-wave (C/D layout: VGPR r, lanes0-15=row r, lanes16-31=row r+8)
  #pragma unroll
  for (int r = 0; r < 8; ++r) {
    float v = p0[r];
    v += __shfl_xor(v, 8, 16);
    v += __shfl_xor(v, 4, 16);
    v += __shfl_xor(v, 2, 16);
    v += __shfl_xor(v, 1, 16);
    p0[r] = v;
    float w = p1[r];
    w += __shfl_xor(w, 8, 16);
    w += __shfl_xor(w, 4, 16);
    w += __shfl_xor(w, 2, 16);
    w += __shfl_xor(w, 1, 16);
    p1[r] = w;
  }
  const float b2v = b2[0];
  const int   ob  = tile * TILE_M;
  if (lane == 0) {
    #pragma unroll
    for (int r = 0; r < 8; ++r) {
      out[ob + r]      = p0[r] + b2v;               // macrotile rows 0..7
      out[ob + 16 + r] = p1[r] + b2v;               // macrotile rows 16..23
    }
  } else if (lane == 16) {
    #pragma unroll
    for (int r = 0; r < 8; ++r) {
      out[ob + 8 + r]  = p0[r] + b2v;               // macrotile rows 8..15
      out[ob + 24 + r] = p1[r] + b2v;               // macrotile rows 24..31
    }
  }
}

extern "C" void kernel_launch(void* const* d_in, const int* in_sizes, int n_in,
                              void* d_out, int out_size, void* d_ws, size_t ws_size,
                              hipStream_t stream) {
  const float* z  = (const float*)d_in[0];
  const int*   ei = (const int*)d_in[1];      // [2, E] int32 (jax default int)
  const float* w1 = (const float*)d_in[2];
  const float* b1 = (const float*)d_in[3];
  const float* w2 = (const float*)d_in[4];
  const float* b2 = (const float*)d_in[5];
  const int E = in_sizes[1] / 2;

  _Float16* frag = (_Float16*)d_ws;           // 256 KB of repacked f16 W1 fragments

  prep_w1<<<(NTILES * KSTEPS * 32 * 16) / 256, 256, 0, stream>>>(w1, frag);

  const int tiles = E / TILE_M;               // 320000/32 = 10000, divisible by WAVES
  edge_mlp<<<tiles / WAVES, BLOCK, 0, stream>>>(z, ei, ei + E, frag,
                                                b1, w2, b2, (float*)d_out);
}
